// ResidualQuantizer_85942295593527
// MI455X (gfx1250) — compile-verified, matrix operand order checked by reading, not run on hardware
//
#include <hip/hip_runtime.h>
#include <hip/hip_bf16.h>

// ---------------------------------------------------------------------------
// VQ quantizer for MI455X (gfx1250, wave32, WMMA bf16, TDM async codebook DMA)
// ---------------------------------------------------------------------------

#define NROWS   65536
#define IN_DIM  512
#define E_DIM   64
#define K_CODES 4096

typedef unsigned short u16;
typedef __attribute__((ext_vector_type(8)))  __bf16 v8bf;
typedef __attribute__((ext_vector_type(16))) __bf16 v16bf;
typedef __attribute__((ext_vector_type(8)))  float  v8f;
typedef __attribute__((ext_vector_type(4)))  unsigned int u32x4;
typedef __attribute__((ext_vector_type(8)))  int i32x8;
typedef __attribute__((ext_vector_type(4)))  int i32x4;

// f32 -> bf16 round-to-nearest-even
static __device__ inline u16 f2bf(float f) {
  unsigned u = __float_as_uint(f);
  unsigned r = 0x7FFFu + ((u >> 16) & 1u);
  return (u16)((u + r) >> 16);
}
static __device__ inline unsigned pack2bf(float a, float b) {
  return (unsigned)f2bf(a) | ((unsigned)f2bf(b) << 16);
}

// Load a 16x32 bf16 fragment (A layout; B uses the symmetric per-lane layout).
// lane 0-15: row = lane, kb = 0 ; lane 16-31: same row, kb = 8.
// Each half (K = kb..kb+7, kb+16..kb+23) is one 16-byte load.
static __device__ inline v16bf load_frag(const u16* base, int stride_elems, int lane) {
  int row = lane & 15;
  int kb  = (lane >> 4) << 3;
  const u16* p = base + row * stride_elems + kb;
  union { v16bf v; v8bf h[2]; } u;
  u.h[0] = *reinterpret_cast<const v8bf*>(p);
  u.h[1] = *reinterpret_cast<const v8bf*>(p + 16);
  return u.v;
}

static __device__ inline v8f wmma_bf16(v16bf a, v16bf b, v8f c) {
  return __builtin_amdgcn_wmma_f32_16x16x32_bf16(false, a, false, b, (short)0, c,
                                                 false, false);
}

// ---------------------------------------------------------------------------
// Tensor Data Mover: flat 1-D tile DMA  global -> LDS  (D# per ISA ch.8)
// ---------------------------------------------------------------------------
#if __has_builtin(__builtin_amdgcn_tensor_load_to_lds)
#define USE_TDM 1
static __device__ inline void tdm_load_lds(const u16* gsrc, u16* ldsdst,
                                           unsigned bytes) {
  // ISA 10.2: LDS generic address truncates to addr[31:0]
  unsigned lds_addr = (unsigned)(size_t)(const void*)ldsdst;
  unsigned long long ga = (unsigned long long)(size_t)gsrc;
  unsigned dw = bytes >> 2;                      // dwords (data_size = 4B)
  // group0: count=1 | lds_addr | global_addr[56:0] | type=2
  u32x4 g0 = { 1u,
               lds_addr,
               (unsigned)ga,
               (unsigned)((ga >> 32) & 0x01FFFFFFu) | 0x80000000u };
  // group1: mask=0, data_size=2(4B); tensor_dim0=tile_dim0=dw; dim1=1; stride=dw
  i32x8 g1;
  g1[0] = (int)(2u << 16);                       // data_size = 4 bytes
  g1[1] = (int)((dw & 0xFFFFu) << 16);           // tensor_dim0[15:0]
  g1[2] = (int)(((dw >> 16) & 0xFFFFu) | (1u << 16)); // tensor_dim0[31:16], dim1=1
  g1[3] = (int)(dw << 16);                       // tile_dim0 (dw <= 0xFFFF)
  g1[4] = 1;                                     // tile_dim1 = 1
  g1[5] = (int)dw;                               // tensor_dim0_stride
  g1[6] = 0;
  g1[7] = 0;
  i32x4 z4 = { 0, 0, 0, 0 };
#if __has_include(<hip/amd_detail/amd_gfx1250_TDM.h>)
  i32x8 z8 = { 0, 0, 0, 0, 0, 0, 0, 0 };
  __builtin_amdgcn_tensor_load_to_lds(g0, g1, z4, z4, z8, 0);   // clang-23 form
#else
  __builtin_amdgcn_tensor_load_to_lds(g0, g1, z4, z4, 0);       // ROCm 7.2 form
#endif
}
#endif

static __device__ inline void stage_copy_256x64(u16* dst_lds, const u16* src, int t) {
  const uint4* s = (const uint4*)src;
  uint4* d = (uint4*)dst_lds;
  #pragma unroll
  for (int i = 0; i < 8; i++) d[t + 256 * i] = s[t + 256 * i];
}

// ---------------------------------------------------------------------------
// Kernel 0: prep — bf16 conversions / transposes, e-norms, zero accumulators
// ---------------------------------------------------------------------------
__global__ void vq_prep_kernel(const float* __restrict__ emb,
                               const float* __restrict__ W_in,
                               const float* __restrict__ W_out,
                               u16* __restrict__ emb_bf,
                               u16* __restrict__ W_inT,
                               u16* __restrict__ W_outT,
                               float* __restrict__ e_norm,
                               unsigned* __restrict__ counts,
                               float* __restrict__ partials) {
  int tid = blockIdx.x * blockDim.x + threadIdx.x;
  if (tid < K_CODES * E_DIM) emb_bf[tid] = f2bf(emb[tid]);
  if (tid < E_DIM * IN_DIM) {                 // W_inT [64][512]
    int n = tid >> 9, k = tid & 511;
    W_inT[tid] = f2bf(W_in[k * E_DIM + n]);
  }
  if (tid < IN_DIM * E_DIM) {                 // W_outT [512][64]
    int n = tid >> 6, k = tid & 63;
    W_outT[tid] = f2bf(W_out[k * IN_DIM + n]);
  }
  if (tid < K_CODES) {
    float s = 0.f;
    #pragma unroll 8
    for (int k = 0; k < E_DIM; k++) { float v = emb[tid * E_DIM + k]; s += v * v; }
    e_norm[tid] = s;
    counts[tid] = 0u;
  }
  if (tid < 512) partials[tid] = 0.f;
}

// ---------------------------------------------------------------------------
// Kernel 1: x = inputs @ W_in + b_in  (store bf16 [N,64])
// block = 256 threads = 8 waves, 32 rows; wave (tileR, colTile) over 2x4
// ---------------------------------------------------------------------------
__global__ void vq_proj_in_kernel(const float* __restrict__ inputs,
                                  const u16* __restrict__ W_inT,
                                  const float* __restrict__ b_in,
                                  u16* __restrict__ x_bf) {
  __shared__ __align__(16) u16 xs[32 * IN_DIM];   // 32 KB bf16 staging
  const int rowbase = blockIdx.x * 32;
  const int t = threadIdx.x;

  { // cooperative load + bf16 convert of 32x512 input tile (contiguous flat)
    const float2* src = (const float2*)(inputs + (size_t)rowbase * IN_DIM) + t * 32;
    unsigned* dst = (unsigned*)xs + t * 32;
    #pragma unroll
    for (int i = 0; i < 32; i++) { float2 v = src[i]; dst[i] = pack2bf(v.x, v.y); }
  }
  __syncthreads();

  const int wave = t >> 5, lane = t & 31;
  const int tileR = wave >> 2;       // 0..1
  const int ct    = wave & 3;        // 0..3 (columns 16*ct .. 16*ct+15)
  const u16* abase = xs + tileR * 16 * IN_DIM;

  v8f c = {};
  #pragma unroll
  for (int kc = 0; kc < IN_DIM / 32; kc++) {
    v16bf a = load_frag(abase + kc * 32, IN_DIM, lane);
    v16bf b = load_frag(W_inT + ct * 16 * IN_DIM + kc * 32, IN_DIM, lane);
    c = wmma_bf16(a, b, c);
  }

  const int n = ct * 16 + (lane & 15);
  const float bias = b_in[n];
  const int mbase = (lane >> 4) * 8;
  #pragma unroll
  for (int r = 0; r < 8; r++) {
    int row = rowbase + tileR * 16 + mbase + r;
    x_bf[(size_t)row * E_DIM + n] = f2bf(c[r] + bias);
  }
}

// ---------------------------------------------------------------------------
// Kernel 2: argmin_j ( ||e_j||^2 - 2 x.e_j )   (||x||^2 dropped — constant)
// block = 256 threads = 8 waves, 128 rows.
// Codebook chunks (256 codes, 32 KB) double-buffered in LDS via TDM DMA:
// wave 0 issues tensor_load_to_lds for chunk cb+1, all waves compute chunk cb,
// s_wait_tensorcnt(0) + barrier at the top of the next iteration.
// ---------------------------------------------------------------------------
__global__ void vq_argmin_kernel(const u16* __restrict__ x_bf,
                                 const u16* __restrict__ emb_bf,
                                 const float* __restrict__ e_norm,
                                 int* __restrict__ idx_out,
                                 unsigned* __restrict__ counts) {
  __shared__ __align__(16) u16 xs[128 * E_DIM];          // 16 KB
  __shared__ __align__(16) u16 ebuf[2][256 * E_DIM];     // 64 KB double buffer
  __shared__ float ens[2][256];
  const int rowbase = blockIdx.x * 128;
  const int t = threadIdx.x;
  const int wave = t >> 5, lane = t & 31;
  const int NCHUNK = K_CODES / 256;

  { // stage x rows (16 KB, contiguous)
    const uint4* src = (const uint4*)(x_bf + (size_t)rowbase * E_DIM);
    uint4* dst = (uint4*)xs;
    #pragma unroll
    for (int i = 0; i < 4; i++) dst[t + 256 * i] = src[t + 256 * i];
  }
  ens[0][t] = e_norm[t];
#ifdef USE_TDM
  if (wave == 0) tdm_load_lds(emb_bf, ebuf[0], 256 * E_DIM * 2);
#else
  stage_copy_256x64(ebuf[0], emb_bf, t);
#endif
  __syncthreads();                       // xs visible to all waves

  const u16* abase = xs + wave * 16 * E_DIM;
  const v16bf a0 = load_frag(abase + 0,  E_DIM, lane);
  const v16bf a1 = load_frag(abase + 32, E_DIM, lane);

  float best[8];
  int   bidx[8];
  #pragma unroll
  for (int r = 0; r < 8; r++) { best[r] = 3.4e38f; bidx[r] = 0x7fffffff; }

  for (int cb = 0; cb < NCHUNK; cb++) {
#ifdef USE_TDM
    if (wave == 0) __builtin_amdgcn_s_wait_tensorcnt(0);  // chunk cb landed
#endif
    __syncthreads();   // chunk cb visible; all reads of other buffer finished

    if (cb + 1 < NCHUNK) {             // stage chunk cb+1 into the other buffer
      ens[(cb + 1) & 1][t] = e_norm[(cb + 1) * 256 + t];
#ifdef USE_TDM
      if (wave == 0)
        tdm_load_lds(emb_bf + (size_t)(cb + 1) * 256 * E_DIM,
                     ebuf[(cb + 1) & 1], 256 * E_DIM * 2);
#else
      stage_copy_256x64(ebuf[(cb + 1) & 1],
                        emb_bf + (size_t)(cb + 1) * 256 * E_DIM, t);
#endif
    }

    const u16* cbuf = ebuf[cb & 1];
    const float* cen = ens[cb & 1];
    #pragma unroll 4
    for (int jt = 0; jt < 16; jt++) {
      const u16* bbase = cbuf + jt * 16 * E_DIM;
      v16bf b0 = load_frag(bbase + 0,  E_DIM, lane);
      v16bf b1 = load_frag(bbase + 32, E_DIM, lane);
      v8f c = {};
      c = wmma_bf16(a0, b0, c);
      c = wmma_bf16(a1, b1, c);
      const int nl = jt * 16 + (lane & 15);
      const float en = cen[nl];
      const int code = cb * 256 + nl;
      #pragma unroll
      for (int r = 0; r < 8; r++) {
        float d = en - 2.0f * c[r];
        if (d < best[r] || (d == best[r] && code < bidx[r])) {
          best[r] = d; bidx[r] = code;
        }
      }
    }
  }

  // reduce argmin across the 16 lanes holding the same row (first-index ties)
  #pragma unroll
  for (int r = 0; r < 8; r++) {
    #pragma unroll
    for (int off = 1; off < 16; off <<= 1) {
      float ov = __shfl_xor(best[r], off, 32);
      int   oi = __shfl_xor(bidx[r], off, 32);
      if (ov < best[r] || (ov == best[r] && oi < bidx[r])) {
        best[r] = ov; bidx[r] = oi;
      }
    }
  }
  if ((lane & 15) == 0) {
    const int mbase = (lane >> 4) * 8;
    #pragma unroll
    for (int r = 0; r < 8; r++) {
      int row = rowbase + wave * 16 + mbase + r;
      idx_out[row] = bidx[r];
      atomicAdd(&counts[bidx[r]], 1u);   // integer atomics: deterministic
    }
  }
}

// ---------------------------------------------------------------------------
// Kernel 3: q = emb[idx] @ W_out + b_out ; write quantized(=q), loss partials
// block = 256 threads = 8 waves, 128 rows
// ---------------------------------------------------------------------------
__global__ void vq_proj_out_kernel(const float* __restrict__ inputs,
                                   const u16* __restrict__ emb_bf,
                                   const u16* __restrict__ W_outT,
                                   const float* __restrict__ b_out,
                                   const int* __restrict__ idx,
                                   float* __restrict__ q_out,
                                   float* __restrict__ partials) {
  __shared__ __align__(16) u16 as[128 * E_DIM];  // 16 KB gathered codebook rows
  __shared__ float wsum[8];
  const int rowbase = blockIdx.x * 128;
  const int t = threadIdx.x;

  { // gather: thread copies half a codebook row (64 bytes)
    int r = t >> 1, half = t & 1;
    int code = idx[rowbase + r];
    const uint4* src = (const uint4*)(emb_bf + (size_t)code * E_DIM + half * 32);
    uint4* dst = (uint4*)(as + r * E_DIM + half * 32);
    #pragma unroll
    for (int i = 0; i < 4; i++) dst[i] = src[i];
  }
  __syncthreads();

  const int wave = t >> 5, lane = t & 31;
  const u16* abase = as + wave * 16 * E_DIM;
  const v16bf a0 = load_frag(abase + 0,  E_DIM, lane);
  const v16bf a1 = load_frag(abase + 32, E_DIM, lane);
  const int nl = lane & 15;
  const int mbase = (lane >> 4) * 8;
  float lsum = 0.f;

  for (int ct = 0; ct < IN_DIM / 16; ct++) {
    const u16* bbase = W_outT + ct * 16 * E_DIM;
    v16bf b0 = load_frag(bbase + 0,  E_DIM, lane);
    v16bf b1 = load_frag(bbase + 32, E_DIM, lane);
    v8f c = {};
    c = wmma_bf16(a0, b0, c);
    c = wmma_bf16(a1, b1, c);
    const int col = ct * 16 + nl;
    const float bias = b_out[col];
    #pragma unroll
    for (int r = 0; r < 8; r++) {
      int row = rowbase + wave * 16 + mbase + r;
      float qv = c[r] + bias;
      q_out[(size_t)row * IN_DIM + col] = qv;           // quantized == q (STE)
      float dv = qv - inputs[(size_t)row * IN_DIM + col];
      lsum += dv * dv;
    }
  }

  #pragma unroll
  for (int off = 1; off < 32; off <<= 1) lsum += __shfl_xor(lsum, off, 32);
  if (lane == 0) wsum[wave] = lsum;
  __syncthreads();
  if (t == 0) {                       // fixed-order block sum -> deterministic
    float s = 0.f;
    #pragma unroll
    for (int i = 0; i < 8; i++) s += wsum[i];
    partials[blockIdx.x] = s;
  }
}

// ---------------------------------------------------------------------------
// Kernel 4: finalize loss + perplexity (fixed-order reductions)
// ---------------------------------------------------------------------------
__global__ void vq_finalize_kernel(const float* __restrict__ partials,
                                   const unsigned* __restrict__ counts,
                                   float* __restrict__ out_scalars) {
  __shared__ float red[256];
  const int t = threadIdx.x;

  float s = partials[t] + partials[t + 256];
  red[t] = s;
  __syncthreads();
  for (int off = 128; off > 0; off >>= 1) {
    if (t < off) red[t] += red[t + off];
    __syncthreads();
  }
  const float total = red[0];
  __syncthreads();

  float h = 0.f;
  for (int i = t; i < K_CODES; i += 256) {
    float p = (float)counts[i] / 65536.0f;
    h += p * logf(p + 1e-10f);
  }
  red[t] = h;
  __syncthreads();
  for (int off = 128; off > 0; off >>= 1) {
    if (t < off) red[t] += red[t + off];
    __syncthreads();
  }
  if (t == 0) {
    out_scalars[0] = 1.25f * total / (65536.0f * 512.0f);  // q_loss + 0.25*e_loss
    out_scalars[1] = expf(-red[0]);                        // perplexity
  }
}

// ---------------------------------------------------------------------------
// launcher
// ---------------------------------------------------------------------------
extern "C" void kernel_launch(void* const* d_in, const int* in_sizes, int n_in,
                              void* d_out, int out_size, void* d_ws, size_t ws_size,
                              hipStream_t stream) {
  const float* inputs = (const float*)d_in[0];
  const float* emb    = (const float*)d_in[1];
  const float* W_in   = (const float*)d_in[2];
  const float* b_in   = (const float*)d_in[3];
  const float* W_out  = (const float*)d_in[4];
  const float* b_out  = (const float*)d_in[5];
  float* out = (float*)d_out;

  char* w = (char*)d_ws;
  u16*      x_bf    = (u16*)(w + 0);            // 8 MB   : x in bf16 [N,64]
  u16*      emb_bf  = (u16*)(w + 8388608);      // 512 KB : emb bf16 [4096,64]
  u16*      W_inT   = (u16*)(w + 8912896);      // 64 KB  : W_in^T bf16 [64,512]
  u16*      W_outT  = (u16*)(w + 8978432);      // 64 KB  : W_out^T bf16 [512,64]
  float*    e_norm  = (float*)(w + 9043968);    // 16 KB
  int*      idxb    = (int*)(w + 9060352);      // 256 KB
  unsigned* counts  = (unsigned*)(w + 9322496); // 16 KB
  float*    partials= (float*)(w + 9338880);    // 2 KB (512 used)

  vq_prep_kernel<<<1024, 256, 0, stream>>>(emb, W_in, W_out, emb_bf, W_inT, W_outT,
                                           e_norm, counts, partials);
  vq_proj_in_kernel<<<NROWS / 32, 256, 0, stream>>>(inputs, W_inT, b_in, x_bf);
  vq_argmin_kernel<<<NROWS / 128, 256, 0, stream>>>(x_bf, emb_bf, e_norm, idxb, counts);
  vq_proj_out_kernel<<<NROWS / 128, 256, 0, stream>>>(inputs, emb_bf, W_outT, b_out,
                                                      idxb, out, partials);
  vq_finalize_kernel<<<1, 256, 0, stream>>>(partials, counts,
                                            out + (size_t)NROWS * IN_DIM);
}
